// CondConv2D_67070209294612
// MI455X (gfx1250) — matrix operand order repaired and checked by reading, not compiled, not run on hardware
//
#include <hip/hip_runtime.h>
#include <hip/hip_bf16.h>

typedef __attribute__((ext_vector_type(16))) __bf16 v16bf;
typedef __attribute__((ext_vector_type(8)))  float  v8f;
typedef __attribute__((ext_vector_type(4)))  float  v4f;

#define B_    16
#define CIN   64
#define H_    256
#define W_    256
#define COUT  64
#define E_    4
#define K_    576   // CIN * 9, reordered as k = (kh*3+kw)*64 + c
#define NST   6     // staging steps: (kh, channel-half)
#define BTROWS 258  // pixels -1..256 (rows 0 and 257 are permanent zero pad)
#define BTPITCH 40  // ushorts per row (80 B, multiple of 16 B)

__device__ __forceinline__ unsigned short f2bf(float f) {
    union { float f; unsigned u; } x; x.f = f;
    unsigned u = x.u;
    unsigned r = (u + 0x7FFFu + ((u >> 16) & 1u)) >> 16;   // RNE
    return (unsigned short)r;
}

// Async global->LDS copy, 16 bytes per lane (tracked by ASYNCcnt).
__device__ __forceinline__ void async_copy16(const unsigned short* g, unsigned short* l) {
    unsigned ldsoff = (unsigned)(unsigned long long)l;
    asm volatile("global_load_async_to_lds_b128 %0, %1, off"
                 :: "v"(ldsoff), "v"(g) : "memory");
}

__device__ __forceinline__ void wait_asynccnt0() {
#if defined(__has_builtin)
#if __has_builtin(__builtin_amdgcn_s_wait_asynccnt)
    __builtin_amdgcn_s_wait_asynccnt(0);
    return;
#endif
#endif
    asm volatile("s_wait_asynccnt 0" ::: "memory");
}

// ---------------------------------------------------------------------------
// Kernel 1: spatial mean per (b,c) + fp32 -> bf16 copy of the input.
// ---------------------------------------------------------------------------
__global__ __launch_bounds__(256) void pool_convert(const float* __restrict__ in,
                                                    unsigned short* __restrict__ inbf,
                                                    float* __restrict__ pooled) {
    const int plane = blockIdx.x;                    // b*CIN + c
    const float* src = in + (size_t)plane * (H_ * W_);
    unsigned short* dst = inbf + (size_t)plane * (H_ * W_);
    const int t = threadIdx.x;

    float sum = 0.f;
    for (int i = t; i < (H_ * W_) / 4; i += 256) {
        v4f v = __builtin_nontemporal_load(((const v4f*)src) + i);  // read-once
        sum += v.x + v.y + v.z + v.w;
        ushort4 o;
        o.x = f2bf(v.x); o.y = f2bf(v.y); o.z = f2bf(v.z); o.w = f2bf(v.w);
        ((ushort4*)dst)[i] = o;
    }
    __shared__ float red[256];
    red[t] = sum;
    __syncthreads();
    for (int s = 128; s > 0; s >>= 1) {
        if (t < s) red[t] += red[t + s];
        __syncthreads();
    }
    if (t == 0) pooled[plane] = red[0] * (1.0f / (H_ * W_));
}

// ---------------------------------------------------------------------------
// Kernel 2: routing = sigmoid(pool @ fc_w^T + fc_b); mix experts -> bf16
// Kmix[b][o][k], k = (kh*3+kw)*64 + c  (tap-major, channel-minor).
// ---------------------------------------------------------------------------
__global__ __launch_bounds__(256) void routing_mix(const float* __restrict__ pooled,
                                                   const float* __restrict__ fcw,
                                                   const float* __restrict__ fcb,
                                                   const float* __restrict__ weight,
                                                   unsigned short* __restrict__ kmix) {
    const int b = blockIdx.x;
    const int t = threadIdx.x;
    __shared__ float pl[CIN];
    __shared__ float r[E_];
    if (t < CIN) pl[t] = pooled[b * CIN + t];
    __syncthreads();
    if (t < E_) {
        float acc = fcb[t];
        for (int c = 0; c < CIN; ++c) acc += pl[c] * fcw[t * CIN + c];
        r[t] = 1.0f / (1.0f + __expf(-acc));
    }
    __syncthreads();
    for (int idx = t; idx < COUT * K_; idx += 256) {
        const int o = idx / K_;
        const int k = idx - o * K_;
        const int pos = k >> 6;        // 0..8 = kh*3+kw
        const int c = k & 63;
        float acc = 0.f;
#pragma unroll
        for (int e = 0; e < E_; ++e)
            acc += r[e] * weight[(((e * COUT + o) * CIN + c) * 9) + pos];
        kmix[(size_t)(b * COUT + o) * K_ + k] = f2bf(acc);
    }
}

// ---------------------------------------------------------------------------
// Kernel 3: implicit-GEMM conv row via WMMA bf16.
// Block = 256 threads (8 waves), computes out[b, 0:64, y, 0:256].
// Stage s = (kh, channel-half): tile Bt[r][kk] = in_bf16[cb+kk][y+kh-1][r-1],
// rows 0 / 257 are zero pad; the kw tap becomes a +kw column offset at the
// B-fragment load (alignment preserved: row pitch 80 B).
// ---------------------------------------------------------------------------
__device__ __forceinline__ void stage_load(const unsigned short* __restrict__ inbf,
                                           int b, int y, int s, uint4 regs[4]) {
    const int kh = s >> 1;
    const int cb = (s & 1) << 5;             // channel base 0 / 32
    const int yy = y + kh - 1;
    const int kk = threadIdx.x & 31;         // channel within half
    const int xq = threadIdx.x >> 5;         // 0..7
    const bool rowok = (yy >= 0) && (yy < H_);
    const unsigned short* src =
        inbf + (((size_t)(b * CIN + cb + kk)) * H_ + (rowok ? yy : 0)) * W_;
#pragma unroll
    for (int j = 0; j < 4; ++j) {
        const int x0 = (xq * 4 + j) * 8;     // aligned 8-pixel group
        regs[j] = rowok ? *(const uint4*)(src + x0) : make_uint4(0, 0, 0, 0);
    }
}

__device__ __forceinline__ void stage_store(unsigned short (*bt)[BTPITCH],
                                            const uint4 regs[4]) {
    const int kk = threadIdx.x & 31;
    const int xq = threadIdx.x >> 5;
#pragma unroll
    for (int j = 0; j < 4; ++j) {
        const int x0 = (xq * 4 + j) * 8;
        const unsigned short* pv = (const unsigned short*)&regs[j];
#pragma unroll
        for (int i = 0; i < 8; ++i)
            bt[x0 + i + 1][kk] = pv[i];      // rows 1..256, unconditional
    }
}

__global__ __launch_bounds__(256) void conv_wmma(const unsigned short* __restrict__ inbf,
                                                 const unsigned short* __restrict__ kmix,
                                                 float* __restrict__ out) {
    __shared__ unsigned short Ak[COUT * K_];              // 72 KB mixed kernel (bf16)
    __shared__ unsigned short Btb[2][BTROWS][BTPITCH];    // ~40 KB double-buffered tile

    const int b = blockIdx.x >> 8;
    const int y = blockIdx.x & 255;
    const int t = threadIdx.x;

    // Stage per-sample kernel matrix into LDS (async global->LDS, b128).
    {
        const unsigned short* ksrc = kmix + (size_t)b * COUT * K_;
        for (int i = t; i < (COUT * K_) / 8; i += 256)
            async_copy16(ksrc + i * 8, &Ak[i * 8]);
        wait_asynccnt0();
    }

    // Permanent zero-pad rows (pixels -1 and 256) in both buffers.
    if (t < 32) {
        Btb[0][0][t] = 0; Btb[0][BTROWS - 1][t] = 0;
        Btb[1][0][t] = 0; Btb[1][BTROWS - 1][t] = 0;
    }

    // Prologue: stage s = 0.
    uint4 regs[4];
    stage_load(inbf, b, y, 0, regs);
    stage_store(Btb[0], regs);
    __syncthreads();

    const int lane  = t & 31;
    const int wv    = t >> 5;         // wave 0..7
    const int mh    = wv & 1;         // M half: C_out tiles mh*2 .. mh*2+1
    const int nq    = wv >> 1;        // N quarter: pixel tiles nq*4 .. nq*4+3
    const int row   = lane & 15;
    const int khalf = lane >> 4;

    v8f acc[2][4];
#pragma unroll
    for (int i = 0; i < 2; ++i)
#pragma unroll
        for (int j = 0; j < 4; ++j)
            acc[i][j] = (v8f){0.f, 0.f, 0.f, 0.f, 0.f, 0.f, 0.f, 0.f};

    for (int s = 0; s < NST; ++s) {
        if (s + 1 < NST) stage_load(inbf, b, y, s + 1, regs);   // overlap global latency

        const int kh = s >> 1, chalf = s & 1;
        unsigned short (*bt)[BTPITCH] = Btb[s & 1];
#pragma unroll
        for (int kw = 0; kw < 3; ++kw) {
            const int kbase = (kh * 3 + kw) * 64 + chalf * 32;  // A column base

            // A fragments (16x32 bf16): lanes 0-15 K0-7,K16-23; 16-31 K8-15,K24-31.
            union { uint4 q[2]; v16bf v; } af[2];
#pragma unroll
            for (int mtl = 0; mtl < 2; ++mtl) {
                const unsigned short* ap =
                    &Ak[((mh * 2 + mtl) * 16 + row) * K_ + kbase + khalf * 8];
                af[mtl].q[0] = *(const uint4*)ap;
                af[mtl].q[1] = *(const uint4*)(ap + 16);
            }
            // B fragments (32x16 bf16): lane = (khalf, n); +kw column shift.
            union { uint4 q[2]; v16bf v; } bf[4];
#pragma unroll
            for (int ntl = 0; ntl < 4; ++ntl) {
                const unsigned short* bp =
                    &bt[(nq * 4 + ntl) * 16 + row + kw][khalf * 8];
                bf[ntl].q[0] = *(const uint4*)bp;
                bf[ntl].q[1] = *(const uint4*)(bp + 16);
            }
#pragma unroll
            for (int mtl = 0; mtl < 2; ++mtl)
#pragma unroll
                for (int ntl = 0; ntl < 4; ++ntl)
                    acc[mtl][ntl] = __builtin_amdgcn_wmma_f32_16x16x32_bf16(
                        false, af[mtl].v, false, bf[ntl].v, (short)0,
                        acc[mtl][ntl], false, false);
        }

        if (s + 1 < NST) stage_store(Btb[(s + 1) & 1], regs);
        __syncthreads();
    }

    // Store D (non-temporal; output is write-once): VGPR i -> M=i (+8 hi lanes), N=row.
    float* ob = out + (((size_t)(b * COUT)) * H_ + y) * W_;
#pragma unroll
    for (int mtl = 0; mtl < 2; ++mtl) {
        const int m0 = (mh * 2 + mtl) * 16 + (khalf ? 8 : 0);
#pragma unroll
        for (int ntl = 0; ntl < 4; ++ntl) {
            const int x = (nq * 4 + ntl) * 16 + row;
#pragma unroll
            for (int i = 0; i < 8; ++i)
                __builtin_nontemporal_store(acc[mtl][ntl][i],
                                            &ob[(size_t)(m0 + i) * (H_ * W_) + x]);
        }
    }
}

// ---------------------------------------------------------------------------
extern "C" void kernel_launch(void* const* d_in, const int* in_sizes, int n_in,
                              void* d_out, int out_size, void* d_ws, size_t ws_size,
                              hipStream_t stream) {
    const float* in     = (const float*)d_in[0];
    const float* weight = (const float*)d_in[1];
    const float* fcw    = (const float*)d_in[2];
    const float* fcb    = (const float*)d_in[3];
    float* out = (float*)d_out;

    unsigned short* inbf = (unsigned short*)d_ws;                         // 128 MB bf16 input
    float* pooled = (float*)((char*)d_ws + (size_t)B_ * CIN * H_ * W_ * 2);
    unsigned short* kmix = (unsigned short*)((char*)pooled + 1024 * sizeof(float));

    pool_convert<<<B_ * CIN, 256, 0, stream>>>(in, inbf, pooled);
    routing_mix<<<B_, 256, 0, stream>>>(pooled, fcw, fcb, weight, kmix);
    conv_wmma<<<B_ * H_, 256, 0, stream>>>(inbf, kmix, out);
}
